// Transformer_50706383897260
// MI455X (gfx1250) — compile-verified, matrix operand order checked by reading, not run on hardware
//
#include <hip/hip_runtime.h>
#include <hip/hip_bf16.h>

// ---------------- problem constants ----------------
constexpr int Bb = 4, S = 2048, D = 1024, H = 4096;
constexpr float SCALE = 0.03125f;        // 1/sqrt(1024)
constexpr float LN_EPS = 1e-5f;
constexpr long long BS = (long long)Bb * S;   // 8192 rows

typedef __attribute__((ext_vector_type(16))) _Float16 v16h;
typedef __attribute__((ext_vector_type(8)))  _Float16 v8h;
typedef __attribute__((ext_vector_type(8)))  float    v8f;

// epilogue flavors
constexpr int EPI_F16          = 0;  // plain f16 store (ctx)
constexpr int EPI_BIAS_F16     = 1;  // + bias[col], f16 store (Q/K/V)
constexpr int EPI_SCALE_F32    = 2;  // * alpha, f32 store (scores)
constexpr int EPI_BIAS_RES_F32 = 3;  // + bias[col] + residual[idx], f32 store
constexpr int EPI_GELU_F16     = 4;  // gelu(x + bias[col]), f16 store (MLP1)

// ---------------- fragment load (NT layout) ----------------
// 16-bit A 16x32 layout: lanes 0-15 row M=lane, K = {0..7,16..23};
// lanes 16-31 same rows, K = {8..15,24..31}.  B (stored [N,K]) is identical.
__device__ __forceinline__ v16h load_frag(const _Float16* __restrict__ p) {
  v8h lo = *(const v8h*)(p);
  v8h hi = *(const v8h*)(p + 16);
  return __builtin_shufflevector(lo, hi, 0,1,2,3,4,5,6,7,8,9,10,11,12,13,14,15);
}

// ---------------- WMMA GEMM: C[M,N] = A[M,K] * B[N,K]^T (+ epilogue) -------
// grid = (N/256, M/128, batch); block = 256 (8 waves, 2x4); wave tile 64x64.
// Per k-step: 16 b128 loads feed 16 v_wmma_f32_16x16x32_f16 (32 FLOP/B).
template<int EPI, typename OutT>
__global__ __launch_bounds__(256)
void gemm_nt(const _Float16* __restrict__ A, const _Float16* __restrict__ B,
             OutT* __restrict__ C, const float* __restrict__ bias,
             const float* __restrict__ res,
             int K, int lda, int ldb, int ldc,
             long long sA, long long sB, long long sC, float alpha) {
  A += (long long)blockIdx.z * sA;
  B += (long long)blockIdx.z * sB;
  const long long cOff = (long long)blockIdx.z * sC;

  const int lane = threadIdx.x & 31;
  const int w    = threadIdx.x >> 5;
  const int m0   = blockIdx.y * 128 + (w >> 2) * 64;
  const int n0   = blockIdx.x * 256 + (w & 3) * 64;
  const int r    = lane & 15;
  const int hi   = lane >> 4;

  v8f acc[4][4] = {};

  const _Float16* ap = A + (long long)(m0 + r) * lda + hi * 8;
  const _Float16* bp = B + (long long)(n0 + r) * ldb + hi * 8;

  for (int k = 0; k < K; k += 32) {
    v16h af[4], bf[4];
#pragma unroll
    for (int i = 0; i < 4; i++) af[i] = load_frag(ap + (long long)i * 16 * lda + k);
#pragma unroll
    for (int j = 0; j < 4; j++) bf[j] = load_frag(bp + (long long)j * 16 * ldb + k);
#pragma unroll
    for (int i = 0; i < 4; i++)
#pragma unroll
      for (int j = 0; j < 4; j++)
        acc[i][j] = __builtin_amdgcn_wmma_f32_16x16x32_f16(
            false, af[i], false, bf[j], (short)0, acc[i][j], false, false);
  }

#pragma unroll
  for (int i = 0; i < 4; i++) {
#pragma unroll
    for (int j = 0; j < 4; j++) {
      const int col = n0 + j * 16 + r;
#pragma unroll
      for (int e = 0; e < 8; e++) {
        const int row = m0 + i * 16 + hi * 8 + e;
        const long long idx = cOff + (long long)row * ldc + col;
        float val = acc[i][j][e];
        if (EPI == EPI_SCALE_F32) val *= alpha;
        if (EPI == EPI_BIAS_F16 || EPI == EPI_BIAS_RES_F32 || EPI == EPI_GELU_F16)
          val += bias[col];
        if (EPI == EPI_GELU_F16)
          val = 0.5f * val * (1.0f + erff(val * 0.70710678118654752f));
        if (EPI == EPI_BIAS_RES_F32) val += res[idx];
        C[idx] = (OutT)val;
      }
    }
  }
}

// ---------------- LayerNorm (f32 in -> f16 out), one block per row ---------
__global__ __launch_bounds__(256)
void ln_f16(const float* __restrict__ x, const float* __restrict__ g,
            const float* __restrict__ be, _Float16* __restrict__ out) {
  const long long row = blockIdx.x;
  const float* xr = x + row * D;
  const int tid = threadIdx.x;
  float v[4], s = 0.f, ss = 0.f;
#pragma unroll
  for (int j = 0; j < 4; j++) {
    v[j] = xr[tid + j * 256];
    s += v[j];
    ss += v[j] * v[j];
  }
  __shared__ float sh[512];
  sh[tid] = s; sh[256 + tid] = ss;
  __syncthreads();
  for (int o = 128; o > 0; o >>= 1) {
    if (tid < o) { sh[tid] += sh[tid + o]; sh[256 + tid] += sh[256 + tid + o]; }
    __syncthreads();
  }
  const float mu   = sh[0] * (1.0f / D);
  const float var  = sh[256] * (1.0f / D) - mu * mu;
  const float rinv = rsqrtf(var + LN_EPS);
#pragma unroll
  for (int j = 0; j < 4; j++) {
    const int i = tid + j * 256;
    out[row * D + i] = (_Float16)((v[j] - mu) * rinv * g[i] + be[i]);
  }
}

// ------- mask (fill 1e-10) + softmax, in-place f32 + f16 copy; block/row ---
__global__ __launch_bounds__(256)
void softmax_mask(float* __restrict__ sc, const int* __restrict__ am,
                  _Float16* __restrict__ p16) {
  const long long row = blockIdx.x;          // 0 .. B*S-1
  const int b = (int)(row / S);
  float* sr = sc + row * S;
  const int* m = am + (long long)b * S;
  const int tid = threadIdx.x;
  float v[8];
  float mx = -3.4e38f;
#pragma unroll
  for (int j = 0; j < 8; j++) {
    const int i = tid + j * 256;
    float t = sr[i];
    t = (m[i] == 0) ? 1e-10f : t;
    v[j] = t;
    mx = fmaxf(mx, t);
  }
  __shared__ float sh[256];
  sh[tid] = mx; __syncthreads();
  for (int o = 128; o > 0; o >>= 1) {
    if (tid < o) sh[tid] = fmaxf(sh[tid], sh[tid + o]);
    __syncthreads();
  }
  mx = sh[0]; __syncthreads();
  float s = 0.f;
#pragma unroll
  for (int j = 0; j < 8; j++) { v[j] = __expf(v[j] - mx); s += v[j]; }
  sh[tid] = s; __syncthreads();
  for (int o = 128; o > 0; o >>= 1) {
    if (tid < o) sh[tid] += sh[tid + o];
    __syncthreads();
  }
  const float inv = 1.0f / sh[0];
#pragma unroll
  for (int j = 0; j < 8; j++) {
    const int i = tid + j * 256;
    const float p = v[j] * inv;
    sr[i] = p;
    p16[row * S + i] = (_Float16)p;
  }
}

// ---------------- f16 tile transpose: V[b,s,d] -> Vt[b,d,s] ----------------
__global__ __launch_bounds__(256)
void transpose16(const _Float16* __restrict__ in, _Float16* __restrict__ out) {
  __shared__ _Float16 t[32][33];
  const int b = blockIdx.z;
  const int s0 = blockIdx.x * 32, d0 = blockIdx.y * 32;
  const _Float16* ib = in + (long long)b * S * D;
  _Float16* ob = out + (long long)b * D * S;
  const int tx = threadIdx.x & 31, ty = threadIdx.x >> 5;   // 32 x 8
#pragma unroll
  for (int j = 0; j < 32; j += 8)
    t[ty + j][tx] = ib[(long long)(s0 + ty + j) * D + d0 + tx];
  __syncthreads();
#pragma unroll
  for (int j = 0; j < 32; j += 8)
    ob[(long long)(d0 + ty + j) * S + s0 + tx] = t[tx][ty + j];
}

// ---------------- f32 -> f16 cast ----------------
__global__ __launch_bounds__(256)
void castf16(const float* __restrict__ in, _Float16* __restrict__ out, int n) {
  const int i = blockIdx.x * 256 + threadIdx.x;
  if (i < n) out[i] = (_Float16)in[i];
}

// ======================= host-side orchestration ===========================
extern "C" void kernel_launch(void* const* d_in, const int* in_sizes, int n_in,
                              void* d_out, int out_size, void* d_ws, size_t ws_size,
                              hipStream_t stream) {
  const float* x   = (const float*)d_in[0];
  const int*   am  = (const int*)  d_in[1];
  const float* Wq  = (const float*)d_in[2];
  const float* bq  = (const float*)d_in[3];
  const float* Wk  = (const float*)d_in[4];
  const float* bk  = (const float*)d_in[5];
  const float* Wv  = (const float*)d_in[6];
  const float* bv  = (const float*)d_in[7];
  const float* Wo  = (const float*)d_in[8];
  const float* bo  = (const float*)d_in[9];
  const float* W1  = (const float*)d_in[10];
  const float* b1  = (const float*)d_in[11];
  const float* W2  = (const float*)d_in[12];
  const float* b2  = (const float*)d_in[13];
  const float* g1  = (const float*)d_in[14];
  const float* be1 = (const float*)d_in[15];
  const float* g2  = (const float*)d_in[16];
  const float* be2 = (const float*)d_in[17];

  float* outp  = (float*)d_out;             // [B*S, D]  (x2, then final out)
  float* probs = outp + BS * D;             // [B, S, S] f32

  // -------- workspace layout (peak 136 MiB, with reuse) --------
  _Float16* Wq16 = (_Float16*)d_ws;
  _Float16* Wk16 = Wq16 + (long long)D * D;
  _Float16* Wv16 = Wk16 + (long long)D * D;
  _Float16* Wo16 = Wv16 + (long long)D * D;
  _Float16* W116 = Wo16 + (long long)D * D;
  _Float16* W216 = W116 + (long long)H * D;
  _Float16* h16  = W216 + (long long)D * H;   // [BS, D]
  _Float16* Q16  = h16  + BS * D;             // [BS, D]
  _Float16* K16  = Q16  + BS * D;             // [BS, D]
  _Float16* V16  = K16  + BS * D;             // [BS, D]
  _Float16* P16  = V16  + BS * D;             // [BS, S]
  _Float16* Vt16 = P16  + BS * S;             // [B, D, S]
  _Float16* ctx16 = h16;                      // reuse (h dead after QKV)
  _Float16* m16   = Q16;                      // reuse (Q dead after scores)
  _Float16* m116  = K16;                      // reuse: K+V+P16 == BS*H elems exactly

  const dim3 blk(256);
  // 1) weight casts
  castf16<<<dim3((D * D) / 256), blk, 0, stream>>>(Wq, Wq16, D * D);
  castf16<<<dim3((D * D) / 256), blk, 0, stream>>>(Wk, Wk16, D * D);
  castf16<<<dim3((D * D) / 256), blk, 0, stream>>>(Wv, Wv16, D * D);
  castf16<<<dim3((D * D) / 256), blk, 0, stream>>>(Wo, Wo16, D * D);
  castf16<<<dim3((H * D) / 256), blk, 0, stream>>>(W1, W116, H * D);
  castf16<<<dim3((D * H) / 256), blk, 0, stream>>>(W2, W216, D * H);

  // 2) LN1
  ln_f16<<<dim3((unsigned)BS), blk, 0, stream>>>(x, g1, be1, h16);

  // 3) Q/K/V projections: [8192,1024] x [1024,1024]^T
  const dim3 gProj(D / 256, (unsigned)(BS / 128), 1);
  gemm_nt<EPI_BIAS_F16, _Float16><<<gProj, blk, 0, stream>>>(
      h16, Wq16, Q16, bq, nullptr, D, D, D, D, 0, 0, 0, 1.0f);
  gemm_nt<EPI_BIAS_F16, _Float16><<<gProj, blk, 0, stream>>>(
      h16, Wk16, K16, bk, nullptr, D, D, D, D, 0, 0, 0, 1.0f);
  gemm_nt<EPI_BIAS_F16, _Float16><<<gProj, blk, 0, stream>>>(
      h16, Wv16, V16, bv, nullptr, D, D, D, D, 0, 0, 0, 1.0f);

  // 4) scores = Q K^T * SCALE  (batched; direct into d_out probs region)
  const dim3 gScores(S / 256, S / 128, Bb);
  gemm_nt<EPI_SCALE_F32, float><<<gScores, blk, 0, stream>>>(
      Q16, K16, probs, nullptr, nullptr, D, D, D, S,
      (long long)S * D, (long long)S * D, (long long)S * S, SCALE);

  // 5) mask + softmax (f32 in-place, f16 copy for next GEMM)
  softmax_mask<<<dim3((unsigned)BS), blk, 0, stream>>>(probs, am, P16);

  // 6) V -> V^T per batch
  transpose16<<<dim3(S / 32, D / 32, Bb), blk, 0, stream>>>(V16, Vt16);

  // 7) ctx = probs @ V   (as NT: probs[S,S] x Vt[D,S]^T)
  const dim3 gCtx(D / 256, S / 128, Bb);
  gemm_nt<EPI_F16, _Float16><<<gCtx, blk, 0, stream>>>(
      P16, Vt16, ctx16, nullptr, nullptr, S, S, S, D,
      (long long)S * S, (long long)D * S, (long long)S * D, 1.0f);

  // 8) x2 = x + ctx @ Wo^T + bo   -> stored in d_out[0 : B*S*D]
  gemm_nt<EPI_BIAS_RES_F32, float><<<gProj, blk, 0, stream>>>(
      ctx16, Wo16, outp, bo, x, D, D, D, D, 0, 0, 0, 1.0f);

  // 9) LN2 on x2
  ln_f16<<<dim3((unsigned)BS), blk, 0, stream>>>(outp, g2, be2, m16);

  // 10) m1 = gelu(m @ W1^T + b1)   [8192,1024] x [4096,1024]^T
  const dim3 gMlp1(H / 256, (unsigned)(BS / 128), 1);
  gemm_nt<EPI_GELU_F16, _Float16><<<gMlp1, blk, 0, stream>>>(
      m16, W116, m116, b1, nullptr, D, D, D, H, 0, 0, 0, 1.0f);

  // 11) out = x2 + m1 @ W2^T + b2   (in-place residual from d_out)
  const dim3 gMlp2(D / 256, (unsigned)(BS / 128), 1);
  gemm_nt<EPI_BIAS_RES_F32, float><<<gMlp2, blk, 0, stream>>>(
      m116, W216, outp, b2, outp, H, H, H, D, 0, 0, 0, 1.0f);
}
